// Seq2SeqSumm_70549132804636
// MI455X (gfx1250) — matrix operand order repaired and checked by reading, not compiled, not run on hardware
//
#include <hip/hip_runtime.h>

#define Bb 128
#define Ss 1024
#define Tt 50
#define Ee 256
#define Hh 256
#define OUTN 128

typedef __attribute__((ext_vector_type(16))) __bf16 bf16x16;
typedef __attribute__((ext_vector_type(8)))  float  f32x8;
typedef __attribute__((ext_vector_type(4)))  int    i32x4;

union frag_u { bf16x16 v; i32x4 q[2]; };

__device__ __forceinline__ unsigned short f2bfu(float f) {
  union { float f; unsigned u; } v; v.f = f;
  unsigned u = v.u;
  return (unsigned short)((u + 0x7FFFu + ((u >> 16) & 1u)) >> 16);
}
__device__ __forceinline__ float bf2f(unsigned short s) {
  union { unsigned u; float f; } v; v.u = ((unsigned)s) << 16;
  return v.f;
}
__device__ __forceinline__ float sigm(float x) { return 1.f / (1.f + __expf(-x)); }

__device__ __forceinline__ f32x8 wmma_bf16(bf16x16 a, bf16x16 b, f32x8 c) {
  // D = A(16x32 bf16) * B(32x16 bf16) + C(16x16 f32)
  return __builtin_amdgcn_wmma_f32_16x16x32_bf16(false, a, false, b, (short)0, c,
                                                 false, false);
}

// A-frag from LDS bf16 row-major [rows][ld]; contiguous k order, element s of lane half h is k = 16h + s.
__device__ __forceinline__ bf16x16 lds_A_frag(const unsigned short* base, int ld, int m0) {
  const int lane = threadIdx.x & 31;
  const unsigned short* p = base + (m0 + (lane & 15)) * ld + ((lane >> 4) << 4);
  frag_u u;
  u.q[0] = *(const i32x4*)p;          // K +0..7
  u.q[1] = *(const i32x4*)(p + 8);    // K +8..15
  return u.v;
}
// B-frag from LDS bf16 [n][ld] (K contiguous per column).
__device__ __forceinline__ bf16x16 lds_B_frag(const unsigned short* base, int ld, int n0) {
  const int lane = threadIdx.x & 31;
  const unsigned short* p = base + (n0 + (lane & 15)) * ld + ((lane >> 4) << 4);
  frag_u u;
  u.q[0] = *(const i32x4*)p;          // 8 consecutive K
  u.q[1] = *(const i32x4*)(p + 8);
  return u.v;
}

// ---------------- generic WMMA GEMM -----------------------------------------
// C[M,N] = act( A1[M,K1]*B1 + (A2[M,K2]*B2) + bias ),  A fp32 (staged->bf16),
// B pre-packed bf16 [N][K]. out1 fp32 (ldc1), optional out2 fp32 (ldc2).
// remap==1: global row r = s*128+b is stored at ((r%128)*Ss + r/128)*ldc1 + n.
// All launches use M % 128 == 0 and N % 64 == 0 (no edge guards needed).
__global__ __launch_bounds__(256) void k_gemm(
    const float* __restrict__ A1, long lda1,
    const unsigned short* __restrict__ B1, int K1,
    const float* __restrict__ A2, long lda2,
    const unsigned short* __restrict__ B2, int K2,
    const float* __restrict__ bias,
    float* __restrict__ out1, long ldc1,
    float* __restrict__ out2, long ldc2,
    int M, int N, int act, int remap)
{
  __shared__ unsigned short As[2][128][40];
  __shared__ unsigned short Bs[2][64][40];
  const int wave = threadIdx.x >> 5, lane = threadIdx.x & 31;
  const long rowBase = (long)blockIdx.x * 128;
  const int nBase = blockIdx.y * 64;

  f32x8 zero = {0.f,0.f,0.f,0.f,0.f,0.f,0.f,0.f};
  f32x8 acc[4];
  acc[0] = zero; acc[1] = zero; acc[2] = zero; acc[3] = zero;

  const int ar = threadIdx.x >> 1;          // A stage: row
  const int ako = (threadIdx.x & 1) << 4;   // A stage: k offset (16 elems)
  const int bn = threadIdx.x >> 2;          // B stage: local n
  const int bko = (threadIdx.x & 3) << 3;   // B stage: k offset (8 elems)

  const int nch1 = K1 >> 5;
  const int nch2 = A2 ? (K2 >> 5) : 0;
  const int nch = nch1 + nch2;

  auto stage = [&](int ci, int buf) {
    const float* A; const unsigned short* Bm; long lda; int K, kc;
    if (ci < nch1) { A = A1; Bm = B1; lda = lda1; K = K1; kc = ci << 5; }
    else           { A = A2; Bm = B2; lda = lda2; K = K2; kc = (ci - nch1) << 5; }
    const float* sa = A + (rowBase + ar) * lda + kc + ako;
    #pragma unroll
    for (int i = 0; i < 16; ++i) As[buf][ar][ako + i] = f2bfu(sa[i]);
    const unsigned short* sb = Bm + (size_t)(nBase + bn) * K + kc + bko;
    #pragma unroll
    for (int i = 0; i < 8; ++i) Bs[buf][bn][bko + i] = sb[i];
  };

  stage(0, 0);
  for (int ci = 0; ci < nch; ++ci) {
    __syncthreads();                       // staged chunk ci visible, prev reads done
    if (ci + 1 < nch) stage(ci + 1, (ci + 1) & 1);
    const unsigned short* Ab = &As[ci & 1][0][0];
    const unsigned short* Bbp = &Bs[ci & 1][0][0];
    // cluster all fragment loads, then issue 4 WMMAs back-to-back
    bf16x16 af = lds_A_frag(Ab, 40, wave * 16);
    bf16x16 b0 = lds_B_frag(Bbp, 40, 0);
    bf16x16 b1 = lds_B_frag(Bbp, 40, 16);
    bf16x16 b2 = lds_B_frag(Bbp, 40, 32);
    bf16x16 b3 = lds_B_frag(Bbp, 40, 48);
    acc[0] = wmma_bf16(af, b0, acc[0]);
    acc[1] = wmma_bf16(af, b1, acc[1]);
    acc[2] = wmma_bf16(af, b2, acc[2]);
    acc[3] = wmma_bf16(af, b3, acc[3]);
  }

  // branch-light epilogue: dims are exact tile multiples.
  const int mrow = wave * 16 + ((lane >> 4) << 3);
  #pragma unroll
  for (int nt = 0; nt < 4; ++nt) {
    const int n = nBase + nt * 16 + (lane & 15);
    const float bv = bias ? bias[n] : 0.f;
    #pragma unroll
    for (int r = 0; r < 8; ++r) {
      const long m = rowBase + mrow + r;
      float v = acc[nt][r] + bv;
      if (act) v = tanhf(v);
      size_t oi;
      if (remap) oi = ((size_t)(m & (Bb - 1)) * Ss + (size_t)(m >> 7)) * (size_t)ldc1 + n;
      else       oi = (size_t)m * (size_t)ldc1 + n;
      out1[oi] = v;
      if (out2) out2[(size_t)m * (size_t)ldc2 + n] = v;
    }
  }
}

// ---------------- persistent bidirectional LSTM encoder ---------------------
__device__ __forceinline__ void grid_barrier(unsigned* cnt, unsigned* gen) {
  __syncthreads();
  if (threadIdx.x == 0) {
    __threadfence();
    unsigned g = __hip_atomic_load(gen, __ATOMIC_ACQUIRE, __HIP_MEMORY_SCOPE_AGENT);
    if (__hip_atomic_fetch_add(cnt, 1u, __ATOMIC_ACQ_REL, __HIP_MEMORY_SCOPE_AGENT) == 15u) {
      __hip_atomic_store(cnt, 0u, __ATOMIC_RELAXED, __HIP_MEMORY_SCOPE_AGENT);
      __hip_atomic_fetch_add(gen, 1u, __ATOMIC_RELEASE, __HIP_MEMORY_SCOPE_AGENT);
    } else {
      while (__hip_atomic_load(gen, __ATOMIC_ACQUIRE, __HIP_MEMORY_SCOPE_AGENT) == g)
        __builtin_amdgcn_s_sleep(2);
    }
  }
  __syncthreads();
}

// grid (16, 2): blockIdx.x = 16-hidden-unit slice, blockIdx.y = direction.
__global__ __launch_bounds__(256) void k_encoder(
    const unsigned short* __restrict__ WhhF, const unsigned short* __restrict__ WhhB,
    const float* __restrict__ XWf, const float* __restrict__ XWb,
    const float* __restrict__ biasF, const float* __restrict__ biasB,
    const int* __restrict__ lens, const float* __restrict__ init_c,
    unsigned short* __restrict__ hbuf,   // [2 dir][2 pp][128][256] bf16
    float* __restrict__ enc_cat,         // [S*B][512] (fwd 0..255, bwd 256..511)
    float* __restrict__ catH, float* __restrict__ catC,  // [128][512]
    unsigned* __restrict__ sync)
{
  const int dir = blockIdx.y;
  const int j0 = blockIdx.x * 16;
  const unsigned short* Wp = dir ? WhhB : WhhF;
  const float* XW = dir ? XWb : XWf;
  const float* bias = dir ? biasB : biasF;
  unsigned* cnt = sync + dir * 2;
  unsigned* gen = cnt + 1;
  unsigned short* hb = hbuf + (size_t)dir * 2 * Bb * Hh;
  const int dirofs = dir ? Hh : 0;

  const int wave = threadIdx.x >> 5, lane = threadIdx.x & 31;
  const int gate = wave & 3;     // i,f,g,o
  const int mhalf = wave >> 2;   // which 4 m-tiles

  // persistent B-fragments: Whh.T columns [gate*H + j0 .. +15], K = 0..255
  bf16x16 bw[8];
  {
    const int gcol = gate * Hh + j0 + (lane & 15);
    const int khi = (lane >> 4) << 4;
    #pragma unroll
    for (int kk = 0; kk < 8; ++kk) {
      const unsigned short* p = Wp + (size_t)gcol * Hh + kk * 32 + khi;
      frag_u u;
      u.q[0] = *(const i32x4*)p;
      u.q[1] = *(const i32x4*)(p + 8);
      bw[kk] = u.v;
    }
  }

  __shared__ float gbuf[128][68];   // gates for this WG's 16 hidden units

  const int cb = threadIdx.x >> 1;          // batch row for cell phase
  const int cj = (threadIdx.x & 1) << 3;    // local j base (8 units/thread)
  float c[8], hlast[8];
  #pragma unroll
  for (int i = 0; i < 8; ++i) { c[i] = init_c[dir * Hh + j0 + cj + i]; hlast[i] = 0.f; }
  const int len = lens[cb];

  for (int s = 0; s < Ss; ++s) {
    const int t = dir ? (Ss - 1 - s) : s;
    const unsigned short* hcur = hb + (size_t)(s & 1) * Bb * Hh;
    unsigned short* hnxt = hb + (size_t)((s & 1) ^ 1) * Bb * Hh;

    f32x8 acc[4];
    #pragma unroll
    for (int mt = 0; mt < 4; ++mt) {
      f32x8 a = {0.f,0.f,0.f,0.f,0.f,0.f,0.f,0.f};
      const int m0 = (mhalf * 4 + mt) * 16;
      const unsigned short* hrow =
          hcur + (size_t)(m0 + (lane & 15)) * Hh + ((lane >> 4) << 4);
      // 4-deep load batches so global loads issue as clauses ahead of WMMAs
      #pragma unroll
      for (int half = 0; half < 2; ++half) {
        frag_u u[4];
        #pragma unroll
        for (int kk = 0; kk < 4; ++kk) {
          const unsigned short* p = hrow + (half * 4 + kk) * 32;
          u[kk].q[0] = *(const i32x4*)p;
          u[kk].q[1] = *(const i32x4*)(p + 8);
        }
        #pragma unroll
        for (int kk = 0; kk < 4; ++kk)
          a = wmma_bf16(u[kk].v, bw[half * 4 + kk], a);
      }
      acc[mt] = a;
    }
    // add precomputed x@Wih.T + bias, scatter into gbuf
    {
      const int jl = lane & 15;
      const int gcol = gate * Hh + j0 + jl;
      const float bsv = bias[gcol];
      #pragma unroll
      for (int mt = 0; mt < 4; ++mt) {
        const int mb = (mhalf * 4 + mt) * 16 + ((lane >> 4) << 3);
        #pragma unroll
        for (int r = 0; r < 8; ++r) {
          const int b = mb + r;
          gbuf[b][gate * 16 + jl] =
              acc[mt][r] + XW[((size_t)b * Ss + t) * 1024 + gcol] + bsv;
        }
      }
    }
    __syncthreads();
    // LSTM cell + pack_padded freezing + ys masking
    #pragma unroll
    for (int i = 0; i < 8; ++i) {
      const int jl = cj + i;
      const int jg = j0 + jl;
      const float ig = sigm(gbuf[cb][jl]);
      const float fg = sigm(gbuf[cb][16 + jl]);
      const float gg = tanhf(gbuf[cb][32 + jl]);
      const float og = sigm(gbuf[cb][48 + jl]);
      const float c2 = fg * c[i] + ig * gg;
      const float h2 = og * tanhf(c2);
      const bool v = (t < len);
      c[i] = v ? c2 : c[i];
      const float hp = bf2f(hcur[(size_t)cb * Hh + jg]);
      const float hn = v ? h2 : hp;
      hlast[i] = hn;
      hnxt[(size_t)cb * Hh + jg] = f2bfu(hn);
      enc_cat[((size_t)t * Bb + cb) * 512 + dirofs + jg] = v ? h2 : 0.f;
    }
    grid_barrier(cnt, gen);
  }
  #pragma unroll
  for (int i = 0; i < 8; ++i) {
    catH[(size_t)cb * 512 + dirofs + j0 + cj + i] = hlast[i];
    catC[(size_t)cb * 512 + dirofs + j0 + cj + i] = c[i];
  }
}

// ---------------- small helper kernels --------------------------------------
__global__ __launch_bounds__(256) void k_convert_direct(const float* __restrict__ s,
                                                        unsigned short* __restrict__ d, int n) {
  const int i = blockIdx.x * 256 + threadIdx.x;
  if (i < n) d[i] = f2bfu(s[i]);
}
__global__ __launch_bounds__(256) void k_convert_transpose(const float* __restrict__ s,
                                                           unsigned short* __restrict__ d,
                                                           int K, int N) {
  const int i = blockIdx.x * 256 + threadIdx.x;
  if (i < K * N) {
    const int n = i / K, k = i % K;
    d[i] = f2bfu(s[(size_t)k * N + n]);
  }
}
__global__ __launch_bounds__(256) void k_bias_sum(const float* __restrict__ a,
                                                  const float* __restrict__ b,
                                                  float* __restrict__ o, int n) {
  const int i = blockIdx.x * 256 + threadIdx.x;
  if (i < n) o[i] = a[i] + b[i];
}
__global__ __launch_bounds__(256) void k_init(const float* __restrict__ ih,
                                              unsigned short* __restrict__ hbuf,
                                              unsigned* __restrict__ sync) {
  const int b = blockIdx.x, dir = blockIdx.y, j = threadIdx.x;
  hbuf[((size_t)dir * 2 * Bb + b) * Hh + j] = f2bfu(ih[dir * Hh + j]);
  if (b == 0 && dir == 0 && j < 4) sync[j] = 0u;
}
__global__ __launch_bounds__(256) void k_seqmean(const float* __restrict__ att,
                                                 const int* __restrict__ lens,
                                                 float* __restrict__ cat2) {
  const int b = blockIdx.x, h = threadIdx.x;
  const float* ab = att + (size_t)b * Ss * Hh + h;
  float sum = 0.f;
  for (int s = 0; s < Ss; ++s) sum += ab[(size_t)s * Hh];
  cat2[b * 512 + Hh + h] = sum / (float)lens[b];
}
__global__ __launch_bounds__(256) void k_cell(const float* __restrict__ gates,
                                              const float* __restrict__ ccur,
                                              float* __restrict__ hnxt,
                                              float* __restrict__ cnxt,
                                              float* __restrict__ cat3,
                                              float* __restrict__ dout, int t) {
  const int b = blockIdx.x, j = threadIdx.x;
  const size_t gb = (size_t)b * 1024;
  const float ig = sigm(gates[gb + j]);
  const float fg = sigm(gates[gb + 256 + j]);
  const float gg = tanhf(gates[gb + 512 + j]);
  const float og = sigm(gates[gb + 768 + j]);
  const float c2 = fg * ccur[b * 256 + j] + ig * gg;
  const float h2 = og * tanhf(c2);
  hnxt[b * 256 + j] = h2;
  cnxt[b * 256 + j] = c2;
  cat3[b * 512 + j] = h2;
  dout[819200 + ((size_t)b * Tt + t) * 256 + j] = h2;    // h_all
  dout[2457600 + ((size_t)b * Tt + t) * 256 + j] = c2;   // c_all
}
// scores + masked softmax + context; one block per batch row, attention in L2.
__global__ __launch_bounds__(256) void k_attn(const float* __restrict__ q,
                                              const float* __restrict__ att,
                                              const int* __restrict__ lens,
                                              float* __restrict__ cat3) {
  __shared__ float qs[256];
  __shared__ float sc[1024];
  __shared__ float red[8];
  const int b = blockIdx.x;
  const int wave = threadIdx.x >> 5, lane = threadIdx.x & 31;
  qs[threadIdx.x] = q[b * 256 + threadIdx.x];
  __syncthreads();
  const float* ab = att + (size_t)b * Ss * Hh;
  const int len = lens[b];
  for (int i = 0; i < 128; ++i) {
    const int s = wave * 128 + i;
    const float* row = ab + (size_t)s * Hh;
    float p = 0.f;
    #pragma unroll
    for (int h = lane; h < 256; h += 32) p += row[h] * qs[h];
    #pragma unroll
    for (int o = 16; o; o >>= 1) p += __shfl_xor(p, o, 32);
    if (lane == 0) sc[s] = (s < len) ? p : -1e30f;
  }
  __syncthreads();
  float m = -3e38f;
  for (int s = threadIdx.x; s < 1024; s += 256) m = fmaxf(m, sc[s]);
  #pragma unroll
  for (int o = 16; o; o >>= 1) m = fmaxf(m, __shfl_xor(m, o, 32));
  if (lane == 0) red[wave] = m;
  __syncthreads();
  if (wave == 0) {
    float mm = (lane < 8) ? red[lane] : -3e38f;
    #pragma unroll
    for (int o = 4; o; o >>= 1) mm = fmaxf(mm, __shfl_xor(mm, o, 32));
    if (lane == 0) red[0] = mm;
  }
  __syncthreads();
  m = red[0];
  __syncthreads();
  float ssum = 0.f;
  for (int s = threadIdx.x; s < 1024; s += 256) {
    const float e = __expf(sc[s] - m);
    sc[s] = e;
    ssum += e;
  }
  #pragma unroll
  for (int o = 16; o; o >>= 1) ssum += __shfl_xor(ssum, o, 32);
  if (lane == 0) red[wave] = ssum;
  __syncthreads();
  if (wave == 0) {
    float t2 = (lane < 8) ? red[lane] : 0.f;
    #pragma unroll
    for (int o = 4; o; o >>= 1) t2 += __shfl_xor(t2, o, 32);
    if (lane == 0) red[0] = t2;
  }
  __syncthreads();
  const float inv = 1.f / red[0];
  float ctx = 0.f;
  const int h = threadIdx.x;
  for (int s = 0; s < 1024; ++s) ctx += sc[s] * ab[(size_t)s * Hh + h];
  cat3[b * 512 + Hh + h] = ctx * inv;
}

// ---------------- host orchestration ----------------------------------------
extern "C" void kernel_launch(void* const* d_in, const int* in_sizes, int n_in,
                              void* d_out, int out_size, void* d_ws, size_t ws_size,
                              hipStream_t stream) {
  (void)in_sizes; (void)n_in; (void)out_size; (void)ws_size;
  const float* article    = (const float*)d_in[0];
  const int*   art_lens   = (const int*)d_in[1];
  const float* target     = (const float*)d_in[2];
  const float* init_enc_h = (const float*)d_in[4];
  const float* init_enc_c = (const float*)d_in[5];
  const float* Wih_f = (const float*)d_in[6];
  const float* Whh_f = (const float*)d_in[7];
  const float* bih_f = (const float*)d_in[8];
  const float* bhh_f = (const float*)d_in[9];
  const float* Wih_b = (const float*)d_in[10];
  const float* Whh_b = (const float*)d_in[11];
  const float* bih_b = (const float*)d_in[12];
  const float* bhh_b = (const float*)d_in[13];
  const float* W_dec_h = (const float*)d_in[14];
  const float* W_dec_c = (const float*)d_in[15];
  const float* attn_wm = (const float*)d_in[16];
  const float* attn_wq = (const float*)d_in[17];
  const float* Wp1 = (const float*)d_in[18];
  const float* bp1 = (const float*)d_in[19];
  const float* Wp2 = (const float*)d_in[20];
  const float* dWih = (const float*)d_in[21];
  const float* dWhh = (const float*)d_in[22];
  const float* dbih = (const float*)d_in[23];
  const float* dbhh = (const float*)d_in[24];
  float* dout = (float*)d_out;

  char* w = (char*)d_ws;
  auto alloc = [&](size_t bytes) -> void* {
    void* p = (void*)w;
    w += (bytes + 255) & ~(size_t)255;
    return p;
  };
  // packed bf16 weights [N][K]
  unsigned short* pWihF = (unsigned short*)alloc(1024 * 256 * 2);
  unsigned short* pWhhF = (unsigned short*)alloc(1024 * 256 * 2);
  unsigned short* pWihB = (unsigned short*)alloc(1024 * 256 * 2);
  unsigned short* pWhhB = (unsigned short*)alloc(1024 * 256 * 2);
  unsigned short* pdWih = (unsigned short*)alloc(1024 * 128 * 2);
  unsigned short* pdWhh = (unsigned short*)alloc(1024 * 256 * 2);
  unsigned short* pWq   = (unsigned short*)alloc(256 * 256 * 2);
  unsigned short* pWm   = (unsigned short*)alloc(256 * 512 * 2);
  unsigned short* pWdh  = (unsigned short*)alloc(256 * 512 * 2);
  unsigned short* pWdc  = (unsigned short*)alloc(256 * 512 * 2);
  unsigned short* pWp1  = (unsigned short*)alloc(256 * 512 * 2);
  unsigned short* pWp2  = (unsigned short*)alloc(128 * 256 * 2);
  unsigned short* hbuf  = (unsigned short*)alloc(2 * 2 * 128 * 256 * 2);
  float* biasF  = (float*)alloc(1024 * 4);
  float* biasB  = (float*)alloc(1024 * 4);
  float* biasD  = (float*)alloc(1024 * 4);
  float* catH   = (float*)alloc(128 * 512 * 4);
  float* catC   = (float*)alloc(128 * 512 * 4);
  float* cat2   = (float*)alloc(128 * 512 * 4);
  float* cat3   = (float*)alloc(128 * 512 * 4);
  float* t1buf  = (float*)alloc(128 * 256 * 4);
  float* xprev0 = (float*)alloc(128 * 128 * 4);
  float* gates  = (float*)alloc(128 * 1024 * 4);
  float* qbuf   = (float*)alloc(128 * 256 * 4);
  float* hdec   = (float*)alloc(2 * 128 * 256 * 4);
  float* cdec   = (float*)alloc(2 * 128 * 256 * 4);
  unsigned* syncp = (unsigned*)alloc(4 * 4);
  float* attention = (float*)alloc((size_t)128 * 1024 * 256 * 4);
  float* enc_cat   = (float*)alloc((size_t)1024 * 128 * 512 * 4);
  float* XWf       = (float*)alloc((size_t)128 * 1024 * 1024 * 4);
  float* XWb       = (float*)alloc((size_t)128 * 1024 * 1024 * 4);

  auto cdirect = [&](const float* s, unsigned short* d, int n) {
    k_convert_direct<<<dim3((n + 255) / 256), dim3(256), 0, stream>>>(s, d, n);
  };
  cdirect(Wih_f, pWihF, 1024 * 256);
  cdirect(Whh_f, pWhhF, 1024 * 256);
  cdirect(Wih_b, pWihB, 1024 * 256);
  cdirect(Whh_b, pWhhB, 1024 * 256);
  cdirect(dWih, pdWih, 1024 * 128);
  cdirect(dWhh, pdWhh, 1024 * 256);
  cdirect(W_dec_h, pWdh, 256 * 512);
  cdirect(W_dec_c, pWdc, 256 * 512);
  cdirect(Wp1, pWp1, 256 * 512);
  cdirect(Wp2, pWp2, 128 * 256);
  k_convert_transpose<<<dim3((256 * 256 + 255) / 256), 256, 0, stream>>>(attn_wq, pWq, 256, 256);
  k_convert_transpose<<<dim3((256 * 512 + 255) / 256), 256, 0, stream>>>(attn_wm, pWm, 512, 256);
  k_bias_sum<<<4, 256, 0, stream>>>(bih_f, bhh_f, biasF, 1024);
  k_bias_sum<<<4, 256, 0, stream>>>(bih_b, bhh_b, biasB, 1024);
  k_bias_sum<<<4, 256, 0, stream>>>(dbih, dbhh, biasD, 1024);
  k_init<<<dim3(128, 2), 256, 0, stream>>>(init_enc_h, hbuf, syncp);

  // precompute x @ Wih.T for all timesteps (big WMMA GEMMs)
  k_gemm<<<dim3(1024, 16), 256, 0, stream>>>(article, 256, pWihF, 256,
      nullptr, 0, nullptr, 0, nullptr, XWf, 1024, nullptr, 0, 131072, 1024, 0, 0);
  k_gemm<<<dim3(1024, 16), 256, 0, stream>>>(article, 256, pWihB, 256,
      nullptr, 0, nullptr, 0, nullptr, XWb, 1024, nullptr, 0, 131072, 1024, 0, 0);

  // persistent recurrent encoder (both directions)
  k_encoder<<<dim3(16, 2), 256, 0, stream>>>(pWhhF, pWhhB, XWf, XWb, biasF, biasB,
      art_lens, init_enc_c, hbuf, enc_cat, catH, catC, syncp);

  // attention = enc_art @ wm  (remapped to [B,S,H])
  k_gemm<<<dim3(1024, 4), 256, 0, stream>>>(enc_cat, 512, pWm, 512,
      nullptr, 0, nullptr, 0, nullptr, attention, 256, nullptr, 0, 131072, 256, 0, 1);
  k_seqmean<<<128, 256, 0, stream>>>(attention, art_lens, cat2);

  // init_h / init_c / init_attn_out
  k_gemm<<<dim3(1, 4), 256, 0, stream>>>(catH, 512, pWdh, 512,
      nullptr, 0, nullptr, 0, nullptr, hdec, 256, cat2, 512, 128, 256, 0, 0);
  k_gemm<<<dim3(1, 4), 256, 0, stream>>>(catC, 512, pWdc, 512,
      nullptr, 0, nullptr, 0, nullptr, cdec, 256, nullptr, 0, 128, 256, 0, 0);
  k_gemm<<<dim3(1, 4), 256, 0, stream>>>(cat2, 512, pWp1, 512,
      nullptr, 0, nullptr, 0, bp1, t1buf, 256, nullptr, 0, 128, 256, 1, 0);
  k_gemm<<<dim3(1, 2), 256, 0, stream>>>(t1buf, 256, pWp2, 256,
      nullptr, 0, nullptr, 0, nullptr, xprev0, 128, nullptr, 0, 128, 128, 0, 0);

  // decoder loop
  for (int t = 0; t < Tt; ++t) {
    const int cur = t & 1, nxt = cur ^ 1;
    const float* xA = (t == 0) ? xprev0 : (target + (size_t)(t - 1) * 128);
    const long xlda = (t == 0) ? 128 : (long)Tt * 128;
    k_gemm<<<dim3(1, 16), 256, 0, stream>>>(xA, xlda, pdWih, 128,
        hdec + cur * 32768, 256, pdWhh, 256, biasD,
        gates, 1024, nullptr, 0, 128, 1024, 0, 0);
    k_cell<<<128, 256, 0, stream>>>(gates, cdec + cur * 32768,
        hdec + nxt * 32768, cdec + nxt * 32768, cat3, dout, t);
    k_gemm<<<dim3(1, 4), 256, 0, stream>>>(hdec + nxt * 32768, 256, pWq, 256,
        nullptr, 0, nullptr, 0, nullptr, qbuf, 256, nullptr, 0, 128, 256, 0, 0);
    k_attn<<<128, 256, 0, stream>>>(qbuf, attention, art_lens, cat3);
    k_gemm<<<dim3(1, 4), 256, 0, stream>>>(cat3, 512, pWp1, 512,
        nullptr, 0, nullptr, 0, bp1, t1buf, 256, nullptr, 0, 128, 256, 1, 0);
    k_gemm<<<dim3(1, 2), 256, 0, stream>>>(t1buf, 256, pWp2, 256,
        nullptr, 0, nullptr, 0, nullptr, dout + t * 128, (long)Tt * 128,
        nullptr, 0, 128, 128, 0, 0);
  }
}
